// SNNActor_84061099917911
// MI455X (gfx1250) — matrix operand order repaired and checked by reading, not compiled, not run on hardware
//
#include <hip/hip_runtime.h>
#include <hip/hip_bf16.h>

typedef _Float16 half16 __attribute__((ext_vector_type(16)));
typedef _Float16 half8  __attribute__((ext_vector_type(8)));
typedef float    float8 __attribute__((ext_vector_type(8)));

#define WMMA_F16(a, b, c) \
  __builtin_amdgcn_wmma_f32_16x16x32_f16(false, (a), false, (b), (short)0, (c), false, false)

constexpr int Bsz = 8192;   // batch
constexpr int F   = 256;    // input features
constexpr int H1  = 256;    // layer1 width
constexpr int H2  = 64;     // layer2 width
constexpr int NA  = 8;      // actions
constexpr int TS  = 20;     // SNN timesteps
constexpr int ROWS = 32;    // batch rows per workgroup (2 row-tiles of 16)

constexpr float BETA = 0.9f;
constexpr float TH   = 1.0f;

// fragment counts (16x16x32 WMMA tiles)
constexpr int NF_W1 = 8 * 16;          // kt x nt
constexpr int NF_W2 = 8 * 4;
constexpr int NF_HD = 2;               // stacked [Wmu;Wsig] head: kt = 0..1
constexpr int FRAG_HALVES = 32 * 16;   // lanes x halves per fragment

// staged-fragment region: W2hi | W2lo | HeadHi | HeadLo (contiguous)
constexpr int SO_W2HI = 0;
constexpr int SO_W2LO = SO_W2HI + NF_W2 * FRAG_HALVES;
constexpr int SO_HDHI = SO_W2LO + NF_W2 * FRAG_HALVES;
constexpr int SO_HDLO = SO_HDHI + NF_HD * FRAG_HALVES;
constexpr int STAGE_HALVES = SO_HDLO + NF_HD * FRAG_HALVES;   // 34816 (68 KB)

// dynamic LDS carve (in halves)
constexpr int L_SPK1 = 0;                        // ROWS*H1 = 8192 halves (16 KB)
constexpr int L_SPK2 = L_SPK1 + ROWS * H1;       // ROWS*H2 = 2048 halves ( 4 KB)
constexpr int L_FRAG = L_SPK2 + ROWS * H2;
constexpr int LDS_HALVES = L_FRAG + STAGE_HALVES;
constexpr size_t LDS_BYTES = LDS_HALVES * sizeof(_Float16);   // 90112 B (88 KB)

// ---------------------------------------------------------------------------
// Pack kernels: build WMMA B-fragments (B = W^T, W is [N][K] row-major f32)
// into fragment-contiguous f16 hi/lo arrays in workspace. Run once; tiny.
// B order used here (k contiguous, as in A): lane n (0-15): col=nBase+n, K=kBase+0..15;
// lanes 16-31: K=kBase+16..31. 16 halves per lane, stored contiguously.
// ---------------------------------------------------------------------------
__device__ inline void write_frag_hilo(_Float16* ph, _Float16* pl,
                                       const float x[16]) {
  half16 hi, lo;
#pragma unroll
  for (int h = 0; h < 16; ++h) {
    _Float16 xh = (_Float16)x[h];
    hi[h] = xh;
    lo[h] = (_Float16)(x[h] - (float)xh);
  }
  *reinterpret_cast<half8*>(ph)     = *(half8*)&hi;
  *reinterpret_cast<half8*>(ph + 8) = *((half8*)&hi + 1);
  *reinterpret_cast<half8*>(pl)     = *(half8*)&lo;
  *reinterpret_cast<half8*>(pl + 8) = *((half8*)&lo + 1);
}

__global__ void pack_bfrags(const float* __restrict__ W, int ldk, int NT,
                            _Float16* __restrict__ dstHi, _Float16* __restrict__ dstLo)
{
  const int f    = blockIdx.x;          // frag index = kt*NT + nt
  const int lane = threadIdx.x;         // 32 threads = 1 wave
  const int kt = f / NT, nt = f % NT;
  const int n  = nt * 16 + (lane & 15);
  const int k0 = kt * 32 + ((lane & 16) ? 16 : 0);

  float x[16];
#pragma unroll
  for (int h = 0; h < 16; ++h) x[h] = W[n * ldk + k0 + h];
  write_frag_hilo(dstHi + (f * 32 + lane) * 16, dstLo + (f * 32 + lane) * 16, x);
}

// Stacked head: cols 0-7 from Wmu, cols 8-15 from Wsig (both [NA][H2]).
__global__ void pack_head_bfrags(const float* __restrict__ Wmu,
                                 const float* __restrict__ Wsig, int ldk,
                                 _Float16* __restrict__ dstHi,
                                 _Float16* __restrict__ dstLo)
{
  const int f    = blockIdx.x;          // kt = 0..1
  const int lane = threadIdx.x;
  const int n    = lane & 15;
  const int k0   = f * 32 + ((lane & 16) ? 16 : 0);
  const float* Wrow = (n < NA) ? (Wmu + n * ldk) : (Wsig + (n - NA) * ldk);

  float x[16];
#pragma unroll
  for (int h = 0; h < 16; ++h) x[h] = Wrow[k0 + h];
  write_frag_hilo(dstHi + (f * 32 + lane) * 16, dstLo + (f * 32 + lane) * 16, x);
}

// ---------------------------------------------------------------------------
// Main kernel helpers
// ---------------------------------------------------------------------------

// Load a packed fragment (two b128 loads straight into the vector halves).
__device__ inline half16 ld_frag(const _Float16* __restrict__ base, int f, int lane) {
  const _Float16* p = base + (f * 32 + lane) * 16;
  half16 r;
  *(half8*)&r       = *reinterpret_cast<const half8*>(p);
  *((half8*)&r + 1) = *reinterpret_cast<const half8*>(p + 8);
  return r;
}

// A-fragment (16x32 f16) from row-major f32 matrix, hi/lo split (one-time use).
__device__ inline void gl_afrag(const float* __restrict__ M, int ld, int rowBase,
                                int kBase, int lane, half16& hi, half16& lo) {
  const int r  = rowBase + (lane & 15);
  const int k0 = kBase + ((lane & 16) ? 16 : 0);
  float x[16];
  const float* p = M + r * ld + k0;
  *(float4*)(x + 0)  = *(const float4*)(p + 0);
  *(float4*)(x + 4)  = *(const float4*)(p + 4);
  *(float4*)(x + 8)  = *(const float4*)(p + 8);
  *(float4*)(x + 12) = *(const float4*)(p + 12);
#pragma unroll
  for (int i = 0; i < 16; ++i) {
    _Float16 h = (_Float16)x[i];
    hi[i] = h;
    lo[i] = (_Float16)(x[i] - (float)h);
  }
}

// A-fragment from f16 spike matrix in LDS (exact {0,1} values).
__device__ inline half16 lds_afrag(const _Float16* S, int ld, int rowBase,
                                   int kBase, int lane) {
  const int r  = rowBase + (lane & 15);
  const int k0 = kBase + ((lane & 16) ? 16 : 0);
  const _Float16* p = S + r * ld + k0;
  half16 r16;
  *(half8*)&r16       = *reinterpret_cast<const half8*>(p);        // ds_load_b128
  *((half8*)&r16 + 1) = *reinterpret_cast<const half8*>(p + 8);
  return r16;
}

// LIF membrane update: reset from pre-update membrane (subtract mechanism).
__device__ inline void lif_update(float8& mem, const float8& z) {
#pragma unroll
  for (int i = 0; i < 8; ++i) {
    float m = mem[i];
    float reset = (m > TH) ? TH : 0.f;
    mem[i] = BETA * m + z[i] - reset;
  }
}

// Spike = (mem > TH) written as f16 {0,1} directly (no f32->f16 convert).
// C layout: n = lane&15, m = v + 8*(lane>=16).
__device__ inline void store_spk(_Float16* S, int ld, int rowBase, int colBase,
                                 int lane, const float8& mem) {
  const int n  = colBase + (lane & 15);
  const int m0 = rowBase + ((lane & 16) ? 8 : 0);
#pragma unroll
  for (int v = 0; v < 8; ++v)
    S[(m0 + v) * ld + n] = (mem[v] > TH) ? (_Float16)1.f : (_Float16)0.f;
}

// ---------------------------------------------------------------------------
// Fused SNN actor kernel (dynamic LDS: spikes + staged weight fragments)
// ---------------------------------------------------------------------------
__global__ __launch_bounds__(256) void snn_actor_fused(
    const float* __restrict__ state,
    const float* __restrict__ b1, const float* __restrict__ b2,
    const float* __restrict__ bmu, const float* __restrict__ bsig,
    const _Float16* __restrict__ fW1hi, const _Float16* __restrict__ fW1lo,
    const _Float16* __restrict__ fStage,   // W2hi | W2lo | HeadHi | HeadLo
    float* __restrict__ out)
{
  extern __shared__ __align__(16) _Float16 lds[];
  _Float16* sSpk1 = lds + L_SPK1;
  _Float16* sSpk2 = lds + L_SPK2;
  _Float16* sFrag = lds + L_FRAG;

  const int lane = threadIdx.x & 31;
  const int wv   = threadIdx.x >> 5;       // 8 waves
  const int l15  = lane & 15;
  const int rowWG = blockIdx.x * ROWS;

  const int rt  = wv >> 2;                 // row tile 0..1 (layer1 + layer2)
  const int ntg = wv & 3;                  // layer1 col-tile group, layer2 n-tile

  // ---- stage loop-invariant weight fragments into LDS (once per WG) ----
  {
    const float4* src = reinterpret_cast<const float4*>(fStage);
    float4*       dst = reinterpret_cast<float4*>(sFrag);
    constexpr int NV4 = STAGE_HALVES * (int)sizeof(_Float16) / 16;  // 4352
#pragma unroll 2
    for (int i = threadIdx.x; i < NV4; i += 256) dst[i] = src[i];
  }

  // ---- z1 = state @ W1^T + b1 (3-term hi/lo split, computed once) ----
  float8 z1[4];
  float8 mem1[4];
#pragma unroll
  for (int t = 0; t < 4; ++t) {
    float bv = b1[(ntg * 4 + t) * 16 + l15];
#pragma unroll
    for (int i = 0; i < 8; ++i) { z1[t][i] = bv; mem1[t][i] = 0.f; }
  }
#pragma unroll
  for (int kt = 0; kt < 8; ++kt) {
    half16 ahi, alo;
    gl_afrag(state, F, rowWG + rt * 16, kt * 32, lane, ahi, alo);
#pragma unroll
    for (int t = 0; t < 4; ++t) {
      const int f = kt * 16 + (ntg * 4 + t);
      half16 bhi = ld_frag(fW1hi, f, lane);
      half16 blo = ld_frag(fW1lo, f, lane);
      z1[t] = WMMA_F16(ahi, bhi, z1[t]);
      z1[t] = WMMA_F16(ahi, blo, z1[t]);
      z1[t] = WMMA_F16(alo, bhi, z1[t]);
    }
  }

  float8 mem2; float8 mem3;
#pragma unroll
  for (int i = 0; i < 8; ++i) { mem2[i] = 0.f; mem3[i] = 0.f; }

  const _Float16* sW2hi = sFrag + SO_W2HI;
  const _Float16* sW2lo = sFrag + SO_W2LO;
  const _Float16* sHdHi = sFrag + SO_HDHI;
  const _Float16* sHdLo = sFrag + SO_HDLO;

  // hoist loop-invariant bias broadcasts
  const float b2v = b2[ntg * 16 + l15];
  const float b3v = (l15 < NA) ? bmu[l15] : bsig[l15 - NA];  // stacked head bias
  const int   rt3 = wv;                                      // waves 0..1 own heads

  // ------------------------------ 20 SNN steps ------------------------------
#pragma unroll 1
  for (int t = 0; t < TS; ++t) {
    // layer1 LIF (membranes in registers), spikes -> LDS as exact f16
#pragma unroll
    for (int tt = 0; tt < 4; ++tt) {
      lif_update(mem1[tt], z1[tt]);
      store_spk(sSpk1, H1, rt * 16, (ntg * 4 + tt) * 16, lane, mem1[tt]);
    }
    __syncthreads();   // also orders one-time fragment staging on iteration 0

    // layer2: z2 = spk1 @ W2^T + b2 (all operands in LDS)
    float8 z2;
#pragma unroll
    for (int i = 0; i < 8; ++i) z2[i] = b2v;
#pragma unroll
    for (int kt = 0; kt < 8; ++kt) {
      half16 a   = lds_afrag(sSpk1, H1, rt * 16, kt * 32, lane);
      half16 bhi = ld_frag(sW2hi, kt * 4 + ntg, lane);
      half16 blo = ld_frag(sW2lo, kt * 4 + ntg, lane);
      z2 = WMMA_F16(a, bhi, z2);
      z2 = WMMA_F16(a, blo, z2);
    }
    lif_update(mem2, z2);
    store_spk(sSpk2, H2, rt * 16, ntg * 16, lane, mem2);
    __syncthreads();

    // layer3: stacked [mu | sigma] head on waves 0..1 (cols 0-7 mu, 8-15 sig)
    if (wv < 2) {
      float8 z3;
#pragma unroll
      for (int i = 0; i < 8; ++i) z3[i] = b3v;
#pragma unroll
      for (int kt = 0; kt < 2; ++kt) {
        half16 a   = lds_afrag(sSpk2, H2, rt3 * 16, kt * 32, lane);
        half16 bhi = ld_frag(sHdHi, kt, lane);
        half16 blo = ld_frag(sHdLo, kt, lane);
        z3 = WMMA_F16(a, bhi, z3);
        z3 = WMMA_F16(a, blo, z3);
      }
      lif_update(mem3, z3);
    }
  }

  // ------------------------------ epilogue ------------------------------
  if (wv < 2) {
    const int m0 = (lane & 16) ? 8 : 0;
    const bool isMu = (l15 < NA);
#pragma unroll
    for (int v = 0; v < 8; ++v) {
      const int row = rowWG + rt3 * 16 + m0 + v;
      const float m = mem3[v];
      float val;
      if (isMu) {
        val = tanhf(m);
        if (isnan(val)) val = 0.f;
      } else {
        float sp = (m > 20.f) ? m : log1pf(expf(m));  // stable softplus
        float s  = sp + 0.03f;
        if (isnan(s)) s = 0.1f;
        val = fminf(fmaxf(s, 0.03f), 1.0f);
      }
      const int col = isMu ? l15 : (l15 - NA);
      out[(isMu ? 0 : Bsz * NA) + row * NA + col] = val;
    }
  }
}

// ---------------------------------------------------------------------------
extern "C" void kernel_launch(void* const* d_in, const int* in_sizes, int n_in,
                              void* d_out, int out_size, void* d_ws, size_t ws_size,
                              hipStream_t stream) {
  const float* state = (const float*)d_in[0];
  const float* W1    = (const float*)d_in[1];
  const float* b1    = (const float*)d_in[2];
  const float* W2    = (const float*)d_in[3];
  const float* b2    = (const float*)d_in[4];
  const float* Wmu   = (const float*)d_in[5];
  const float* bmu   = (const float*)d_in[6];
  const float* Wsig  = (const float*)d_in[7];
  const float* bsig  = (const float*)d_in[8];
  float* out = (float*)d_out;

  // workspace layout (f16 halves): W1 frags, then the contiguous staged region
  _Float16* ws     = (_Float16*)d_ws;
  _Float16* fW1hi  = ws;
  _Float16* fW1lo  = fW1hi + NF_W1 * FRAG_HALVES;
  _Float16* fStage = fW1lo + NF_W1 * FRAG_HALVES;      // W2hi..HeadLo
  _Float16* fW2hi  = fStage + SO_W2HI;
  _Float16* fW2lo  = fStage + SO_W2LO;
  _Float16* fHdHi  = fStage + SO_HDHI;
  _Float16* fHdLo  = fStage + SO_HDLO;

  // allow > 64 KB dynamic LDS (CDNA5: up to 320 KB per workgroup)
  static_assert(LDS_BYTES <= 320 * 1024, "LDS budget");
  hipFuncSetAttribute(reinterpret_cast<const void*>(snn_actor_fused),
                      hipFuncAttributeMaxDynamicSharedMemorySize, (int)LDS_BYTES);

  // one-shot weight packing (tiny; L2-resident afterwards)
  hipLaunchKernelGGL(pack_bfrags, dim3(NF_W1), dim3(32), 0, stream, W1, F,  16, fW1hi, fW1lo);
  hipLaunchKernelGGL(pack_bfrags, dim3(NF_W2), dim3(32), 0, stream, W2, H1, 4,  fW2hi, fW2lo);
  hipLaunchKernelGGL(pack_head_bfrags, dim3(NF_HD), dim3(32), 0, stream,
                     Wmu, Wsig, H2, fHdHi, fHdLo);

  hipLaunchKernelGGL(snn_actor_fused, dim3(Bsz / ROWS), dim3(256), LDS_BYTES, stream,
                     state, b1, b2, bmu, bsig, fW1hi, fW1lo, fStage, out);
}
